// Attention_75307956568620
// MI455X (gfx1250) — matrix operand order repaired and checked by reading, not compiled, not run on hardware
//
#include <hip/hip_runtime.h>

typedef __attribute__((ext_vector_type(16))) __bf16 v16bf;
typedef __attribute__((ext_vector_type(8)))  __bf16 v8bf;
typedef __attribute__((ext_vector_type(8)))  float  v8f;

#define BATCH  4
#define SEQ    2048
#define DMODEL 1024
#define HEADS  16
#define DHEAD  64
#define NC3    3072
#define MTOT   (BATCH*SEQ)   // 8192

// ---------------------------------------------------------------- helpers ---

__device__ __forceinline__ v8f wmma_bf16(v16bf a, v16bf b, v8f c) {
  // (neg_a, A, neg_b, B, c_mod, C, reuse_a, reuse_b)
  return __builtin_amdgcn_wmma_f32_16x16x32_bf16(false, a, false, b,
                                                 (short)0, c, false, false);
}

// A-fragment (16x32 bf16): per lane two 8-element contiguous chunks.
__device__ __forceinline__ v16bf ldA_bf16(const __bf16* p0, const __bf16* p1) {
  v8bf lo = *(const v8bf*)p0;
  v8bf hi = *(const v8bf*)p1;
  return __builtin_shufflevector(lo, hi, 0,1,2,3,4,5,6,7,8,9,10,11,12,13,14,15);
}

// --------------------------------------------------------- bf16 pre-passes --

__global__ void cvt_x(const float* __restrict__ x, __bf16* __restrict__ xb) {
  int idx = blockIdx.x * 256 + threadIdx.x;       // 8192*1024 elements
  xb[idx] = (__bf16)x[idx];
}

__global__ void cvt_wqkv(const float* __restrict__ w, __bf16* __restrict__ wt) {
  int idx = blockIdx.x * 256 + threadIdx.x;       // [1024 x 3072] row-major
  int k = idx / NC3, n = idx % NC3;
  wt[(size_t)n * DMODEL + k] = (__bf16)w[idx];    // transposed: [n][k]
}

__global__ void cvt_wout(const float* __restrict__ w, __bf16* __restrict__ wt) {
  int idx = blockIdx.x * 256 + threadIdx.x;       // [1024 x 1024] row-major
  int k = idx / DMODEL, n = idx % DMODEL;
  wt[(size_t)n * DMODEL + k] = (__bf16)w[idx];    // transposed: [n][k]
}

// ------------------------------------------------------------- QKV GEMM -----
// C[8192,3072] = x @ W_qkv ; each wave owns a 32x32 C tile (4 WMMAs / K-step).
// Scatter into q (scaled, [bh][n][d]), k ([bh][n][d]), v^T ([bh][d][n]).

__global__ void __launch_bounds__(256)
gemm_qkv(const __bf16* __restrict__ xb, const __bf16* __restrict__ wT,
         __bf16* __restrict__ qb, __bf16* __restrict__ kb,
         __bf16* __restrict__ vt) {
  const int wave = threadIdx.x >> 5;
  const int lane = threadIdx.x & 31;
  const int half = lane >> 4;
  const int l16  = lane & 15;
  const int mw = wave >> 2, nw = wave & 3;          // 2 x 4 wave grid
  const int mbase = blockIdx.y * 64 + mw * 32;      // block: 64(M) x 128(N)
  const int nbase = blockIdx.x * 128 + nw * 32;

  const int ka = half ? 16 : 0;                     // A k-offset per lane-half
  const int kbo = half ? 16 : 0;                    // B k-offset per lane-half
  const __bf16* a0 = xb + (size_t)(mbase + l16) * DMODEL + ka;
  const __bf16* a1 = xb + (size_t)(mbase + 16 + l16) * DMODEL + ka;
  const __bf16* b0 = wT + (size_t)(nbase + l16) * DMODEL + kbo;
  const __bf16* b1 = wT + (size_t)(nbase + 16 + l16) * DMODEL + kbo;

  v8f acc[2][2] = {};
#pragma unroll 4
  for (int k = 0; k < DMODEL; k += 32) {
    if (k + 256 < DMODEL) {
      __builtin_prefetch(a0 + k + 256, 0, 3);
      __builtin_prefetch(b0 + k + 256, 0, 3);
    }
    v16bf af0 = ldA_bf16(a0 + k, a0 + k + 8);
    v16bf af1 = ldA_bf16(a1 + k, a1 + k + 8);
    v16bf bf0 = *(const v16bf*)(b0 + k);
    v16bf bf1 = *(const v16bf*)(b1 + k);
    acc[0][0] = wmma_bf16(af0, bf0, acc[0][0]);
    acc[0][1] = wmma_bf16(af0, bf1, acc[0][1]);
    acc[1][0] = wmma_bf16(af1, bf0, acc[1][0]);
    acc[1][1] = wmma_bf16(af1, bf1, acc[1][1]);
  }

#pragma unroll
  for (int nt = 0; nt < 2; ++nt) {
    const int nc  = nbase + nt * 16 + l16;
    const int sec = nc >> 10, cc = nc & 1023;
    const int h = cc >> 6, dd = cc & 63;
#pragma unroll
    for (int mt = 0; mt < 2; ++mt) {
#pragma unroll
      for (int r = 0; r < 8; ++r) {
        const int m = mbase + mt * 16 + r + (half ? 8 : 0);
        const int bi = m >> 11, nrow = m & 2047;
        const size_t bh = (size_t)bi * HEADS + h;
        const float val = acc[mt][nt][r];
        if (sec == 0)      qb[(bh * SEQ + nrow) * DHEAD + dd] = (__bf16)(val * 0.125f);
        else if (sec == 1) kb[(bh * SEQ + nrow) * DHEAD + dd] = (__bf16)val;
        else               vt[(bh * DHEAD + dd) * SEQ + nrow] = (__bf16)val;
      }
    }
  }
}

// --------------------------------------------------- flash attention (fwd) --
// One wave = 16 query rows; 32-key tiles; online softmax; causal mask.

__global__ void __launch_bounds__(256)
attn_fwd(const __bf16* __restrict__ qb, const __bf16* __restrict__ kb,
         const __bf16* __restrict__ vt, __bf16* __restrict__ aout) {
  __shared__ __bf16 Plds[8][16][32];   // per-wave P tile (C-layout -> A-layout)

  const int bh = blockIdx.x;
  const int bi = bh >> 4, h = bh & 15;
  const int wave = threadIdx.x >> 5;
  const int lane = threadIdx.x & 31;
  const int half = lane >> 4, l16 = lane & 15;
  const int R = blockIdx.y * 128 + wave * 16;

  const __bf16* q   = qb + (size_t)bh * SEQ * DHEAD;
  const __bf16* kkp = kb + (size_t)bh * SEQ * DHEAD;
  const __bf16* vvp = vt + (size_t)bh * DHEAD * SEQ;

  const int koff = half ? 16 : 0;
  const size_t qoff = (size_t)(R + l16) * DHEAD;
  v16bf aq0 = ldA_bf16(q + qoff + koff,      q + qoff + koff + 8);
  v16bf aq1 = ldA_bf16(q + qoff + 32 + koff, q + qoff + 32 + koff + 8);

  float mrow[8], lrow[8];
#pragma unroll
  for (int r = 0; r < 8; ++r) { mrow[r] = -1e30f; lrow[r] = 0.f; }
  v8f o0 = {}, o1 = {}, o2 = {}, o3 = {};

  for (int jb = 0; jb <= R + 15; jb += 32) {
    // S = Q @ K^T : two 16x16 C tiles, contraction over d=64 in 2 chunks
    v8f s0 = {}, s1 = {};
    {
      const __bf16* kb0 = kkp + (size_t)(jb + l16) * DHEAD + (half ? 16 : 0);
      const __bf16* kb1 = kkp + (size_t)(jb + 16 + l16) * DHEAD + (half ? 16 : 0);
      s0 = wmma_bf16(aq0, *(const v16bf*)(kb0),      s0);
      s0 = wmma_bf16(aq1, *(const v16bf*)(kb0 + 32), s0);
      s1 = wmma_bf16(aq0, *(const v16bf*)(kb1),      s1);
      s1 = wmma_bf16(aq1, *(const v16bf*)(kb1 + 32), s1);
    }

    const int key0 = jb + l16;
    const int key1 = jb + 16 + l16;
#pragma unroll
    for (int r = 0; r < 8; ++r) {
      const int m = R + r + (half ? 8 : 0);
      float v0 = (key0 > m) ? -1e30f : s0[r];     // causal mask
      float v1 = (key1 > m) ? -1e30f : s1[r];
      float mt = fmaxf(v0, v1);                   // row max across 16 lanes
      mt = fmaxf(mt, __shfl_xor(mt, 1, 16));
      mt = fmaxf(mt, __shfl_xor(mt, 2, 16));
      mt = fmaxf(mt, __shfl_xor(mt, 4, 16));
      mt = fmaxf(mt, __shfl_xor(mt, 8, 16));
      const float nm = fmaxf(mrow[r], mt);
      const float sc = __expf(mrow[r] - nm);
      const float e0 = __expf(v0 - nm);
      const float e1 = __expf(v1 - nm);
      float rs = e0 + e1;
      rs += __shfl_xor(rs, 1, 16);
      rs += __shfl_xor(rs, 2, 16);
      rs += __shfl_xor(rs, 4, 16);
      rs += __shfl_xor(rs, 8, 16);
      lrow[r] = lrow[r] * sc + rs;
      mrow[r] = nm;
      o0[r] *= sc; o1[r] *= sc; o2[r] *= sc; o3[r] *= sc;
      const int prow = r + (half ? 8 : 0);
      Plds[wave][prow][l16]      = (__bf16)e0;
      Plds[wave][prow][16 + l16] = (__bf16)e1;
    }

    // P (16x32) as A-fragment from LDS; V^T rows give contiguous B-fragments
    v16bf ap = ldA_bf16(&Plds[wave][l16][koff], &Plds[wave][l16][koff + 8]);
    const __bf16* vb = vvp + jb + (half ? 16 : 0);
    o0 = wmma_bf16(ap, *(const v16bf*)(vb + (size_t)(l16     ) * SEQ), o0);
    o1 = wmma_bf16(ap, *(const v16bf*)(vb + (size_t)(16 + l16) * SEQ), o1);
    o2 = wmma_bf16(ap, *(const v16bf*)(vb + (size_t)(32 + l16) * SEQ), o2);
    o3 = wmma_bf16(ap, *(const v16bf*)(vb + (size_t)(48 + l16) * SEQ), o3);
  }

#pragma unroll
  for (int r = 0; r < 8; ++r) {
    const int rr = r + (half ? 8 : 0);
    const int n  = R + rr;
    const float inv = 1.f / lrow[r];
    const size_t base = ((size_t)bi * SEQ + n) * DMODEL + h * DHEAD;
    aout[base +      l16] = (__bf16)(o0[r] * inv);
    aout[base + 16 + l16] = (__bf16)(o1[r] * inv);
    aout[base + 32 + l16] = (__bf16)(o2[r] * inv);
    aout[base + 48 + l16] = (__bf16)(o3[r] * inv);
  }
}

// ------------------------------------------------------------ output GEMM ---
// out[8192,1024] = attn @ W_out + b ; 32x32 C tile per wave.

__global__ void __launch_bounds__(256)
gemm_out(const __bf16* __restrict__ a, const __bf16* __restrict__ wT,
         const float* __restrict__ bias, float* __restrict__ out) {
  const int wave = threadIdx.x >> 5;
  const int lane = threadIdx.x & 31;
  const int half = lane >> 4, l16 = lane & 15;
  const int mw = wave >> 2, nw = wave & 3;
  const int mbase = blockIdx.y * 64 + mw * 32;
  const int nbase = blockIdx.x * 128 + nw * 32;

  const int ka = half ? 16 : 0;
  const int kbo = half ? 16 : 0;
  const __bf16* a0 = a  + (size_t)(mbase + l16) * DMODEL + ka;
  const __bf16* a1 = a  + (size_t)(mbase + 16 + l16) * DMODEL + ka;
  const __bf16* b0 = wT + (size_t)(nbase + l16) * DMODEL + kbo;
  const __bf16* b1 = wT + (size_t)(nbase + 16 + l16) * DMODEL + kbo;

  v8f acc[2][2] = {};
#pragma unroll 4
  for (int k = 0; k < DMODEL; k += 32) {
    if (k + 256 < DMODEL) {
      __builtin_prefetch(a0 + k + 256, 0, 3);
      __builtin_prefetch(b0 + k + 256, 0, 3);
    }
    v16bf af0 = ldA_bf16(a0 + k, a0 + k + 8);
    v16bf af1 = ldA_bf16(a1 + k, a1 + k + 8);
    v16bf bf0 = *(const v16bf*)(b0 + k);
    v16bf bf1 = *(const v16bf*)(b1 + k);
    acc[0][0] = wmma_bf16(af0, bf0, acc[0][0]);
    acc[0][1] = wmma_bf16(af0, bf1, acc[0][1]);
    acc[1][0] = wmma_bf16(af1, bf0, acc[1][0]);
    acc[1][1] = wmma_bf16(af1, bf1, acc[1][1]);
  }

#pragma unroll
  for (int nt = 0; nt < 2; ++nt) {
    const int nc = nbase + nt * 16 + l16;
    const float bv = bias[nc];
#pragma unroll
    for (int mt = 0; mt < 2; ++mt) {
#pragma unroll
      for (int r = 0; r < 8; ++r) {
        const int m = mbase + mt * 16 + r + (half ? 8 : 0);
        out[(size_t)m * DMODEL + nc] = acc[mt][nt][r] + bv;
      }
    }
  }
}

// --------------------------------------------------------------- launcher ---

extern "C" void kernel_launch(void* const* d_in, const int* in_sizes, int n_in,
                              void* d_out, int out_size, void* d_ws, size_t ws_size,
                              hipStream_t stream) {
  const float* x    = (const float*)d_in[0];
  // d_in[1] = padding mask: all-valid in this problem setup (unused)
  const float* Wqkv = (const float*)d_in[2];
  const float* Wout = (const float*)d_in[3];
  const float* bout = (const float*)d_in[4];
  float* out = (float*)d_out;

  char* ws = (char*)d_ws;
  __bf16* wqkvT = (__bf16*)(ws);                 //  6 MB  [3072][1024]
  __bf16* woutT = (__bf16*)(ws + 6291456);       //  2 MB  [1024][1024]
  __bf16* xb    = (__bf16*)(ws + 8388608);       // 16 MB  [8192][1024]
  __bf16* qb    = (__bf16*)(ws + 25165824);      // 16 MB  [64][2048][64]
  __bf16* kbuf  = (__bf16*)(ws + 41943040);      // 16 MB  [64][2048][64]
  __bf16* vtb   = (__bf16*)(ws + 58720256);      // 16 MB  [64][64][2048]
  __bf16* aout  = (__bf16*)(ws + 75497472);      // 16 MB  [8192][1024]

  cvt_x<<<32768, 256, 0, stream>>>(x, xb);           // 8192*1024 / 256
  cvt_wqkv<<<12288, 256, 0, stream>>>(Wqkv, wqkvT);  // 1024*3072 / 256
  cvt_wout<<<4096, 256, 0, stream>>>(Wout, woutT);   // 1024*1024 / 256
  gemm_qkv<<<dim3(24, 128), 256, 0, stream>>>(xb, wqkvT, qb, kbuf, vtb);
  attn_fwd<<<dim3(64, 16), 256, 0, stream>>>(qb, kbuf, vtb, aout);
  gemm_out<<<dim3(8, 128), 256, 0, stream>>>(aout, woutT, bout, out);
}
